// HGConvLayer_47493748359389
// MI455X (gfx1250) — matrix operand order repaired and checked by reading, not compiled, not run on hardware
//
#include <hip/hip_runtime.h>
#include <cstdint>
#include <cstddef>

// ---------------- problem constants (match reference) ----------------
#define KHEADS   8
#define DHEAD    32
#define KD       256
#define IN_DIM   256
#define N_AUTHOR 50000
#define N_PAPER  50000
#define NEDGE    400000
#define SLOPE    0.2f

typedef __attribute__((ext_vector_type(16))) __bf16 v16bf;
typedef __attribute__((ext_vector_type(8)))  float  v8f;

__device__ __forceinline__ float lrelu(float x) { return x > 0.0f ? x : SLOPE * x; }

// =====================================================================
// GEMM: Y[N,256] = act(X)[N,256] @ W[256,256]  (+bias) (+gated blend)
// Block = 256 threads = 8 waves. Block tile = 64 rows x 256 cols.
// Wave w owns col tiles [w*32, w*32+32) (2x 16-wide WMMA tiles) x all
// 4 row tiles -> 8 v8f accumulators.
// Double-buffered LDS: per K-step, next slice's global b128 loads are
// issued BEFORE the 8 WMMAs so s_wait_loadcnt lands after the matrix
// work; cvt+ds_store into the other buffer; one barrier per step.
// W slice staged transposed ([col][k]) so B fragment is K = h + 8*g + 8*(h>>3);
// X staged [row][k]; layouts per ISA 7.12.2.
// =====================================================================
__global__ __launch_bounds__(256) void gemm_n256(
    const float* __restrict__ X, const float* __restrict__ W,
    const float* __restrict__ bias,   // nullable: add bias[col]
    const float* __restrict__ gate,   // nullable: scalar, sigmoid gate
    const float* __restrict__ resid,  // nullable: y = g*y + (1-g)*resid
    float* __restrict__ Y, int N, int inputRelu)
{
    __shared__ __bf16 Xs[2][64][34];    // [buf][row][k], padded
    __shared__ __bf16 Wt[2][256][34];   // [buf][col][k] (transposed W slice)

    const int t    = threadIdx.x;
    const int wave = t >> 5;
    const int lane = t & 31;
    const int mrow = lane & 15;      // M (A) or N (B, C/D) within a tile
    const int g    = lane >> 4;      // lane half-group
    const int row0 = blockIdx.x * 64;

    v8f acc[4][2];
    #pragma unroll
    for (int r = 0; r < 4; ++r)
        #pragma unroll
        for (int c = 0; c < 2; ++c)
            #pragma unroll
            for (int j = 0; j < 8; ++j) acc[r][c][j] = 0.0f;

    // ---- fixed per-thread staging addresses ----
    // X: 64 rows x 32 k as 512 float4; thread covers vi = t and t+256
    const int xr0 = t >> 3;              // rows 0..31
    const int xr1 = xr0 + 32;            // rows 32..63
    const int xc  = (t & 7) * 4;         // k-quad
    const int gr0 = min(row0 + xr0, N - 1);   // clamped (store-guarded later)
    const int gr1 = min(row0 + xr1, N - 1);
    // W: 32 k x 256 cols as 2048 float4; thread covers vi = t + i*256
    const int wk0 = t >> 6;              // base k (step +4 per i)
    const int wcg = (t & 63) * 4;        // col-quad

    float4 xreg[2];
    float4 wreg[8];

    auto loadSlice = [&](int ks) {
        xreg[0] = *(const float4*)(X + (size_t)gr0 * 256 + ks + xc);
        xreg[1] = *(const float4*)(X + (size_t)gr1 * 256 + ks + xc);
        #pragma unroll
        for (int i = 0; i < 8; ++i)
            wreg[i] = *(const float4*)(W + (size_t)(ks + wk0 + i * 4) * 256 + wcg);
    };
    auto storeSlice = [&](int b) {
        float vx[8] = { xreg[0].x, xreg[0].y, xreg[0].z, xreg[0].w,
                        xreg[1].x, xreg[1].y, xreg[1].z, xreg[1].w };
        if (inputRelu) {
            #pragma unroll
            for (int j = 0; j < 8; ++j) vx[j] = fmaxf(vx[j], 0.0f);
        }
        #pragma unroll
        for (int j = 0; j < 4; ++j) {
            Xs[b][xr0][xc + j] = (__bf16)vx[j];
            Xs[b][xr1][xc + j] = (__bf16)vx[4 + j];
        }
        #pragma unroll
        for (int i = 0; i < 8; ++i) {
            Wt[b][wcg + 0][wk0 + i * 4] = (__bf16)wreg[i].x;
            Wt[b][wcg + 1][wk0 + i * 4] = (__bf16)wreg[i].y;
            Wt[b][wcg + 2][wk0 + i * 4] = (__bf16)wreg[i].z;
            Wt[b][wcg + 3][wk0 + i * 4] = (__bf16)wreg[i].w;
        }
    };
    auto compute = [&](int b) {
        v16bf bfrag[2];
        #pragma unroll
        for (int c = 0; c < 2; ++c) {
            int col = wave * 32 + c * 16 + mrow;
            #pragma unroll
            for (int h = 0; h < 16; ++h)
                bfrag[c][h] = Wt[b][col][h + 8 * g + ((h >> 3) << 3)];
        }
        #pragma unroll
        for (int r = 0; r < 4; ++r) {
            v16bf afrag;    // ISA 7.12.2 16-bit A layout
            #pragma unroll
            for (int h = 0; h < 16; ++h) {
                int v = h >> 1, lo = h & 1;
                int kk = (v < 4) ? (2 * v + lo + 8 * g)
                                 : (16 + 2 * (v - 4) + lo + 8 * g);
                afrag[h] = Xs[b][r * 16 + mrow][kk];
            }
            acc[r][0] = __builtin_amdgcn_wmma_f32_16x16x32_bf16(
                false, afrag, false, bfrag[0], (short)0, acc[r][0], false, false);
            acc[r][1] = __builtin_amdgcn_wmma_f32_16x16x32_bf16(
                false, afrag, false, bfrag[1], (short)0, acc[r][1], false, false);
        }
    };

    // prologue: fill buffer 0
    loadSlice(0);
    storeSlice(0);
    __syncthreads();

    #pragma unroll
    for (int step = 0; step < 8; ++step) {
        const int cur = step & 1;
        if (step < 7) loadSlice((step + 1) * 32);          // overlap w/ WMMA
        if (step < 6) __builtin_prefetch(W + (size_t)(step + 2) * 32 * 256 + t, 0, 1);
        compute(cur);
        if (step < 7) storeSlice(cur ^ 1);
        __syncthreads();
    }

    float gv = 1.0f, gc = 0.0f;
    if (resid) {
        float gg = 1.0f / (1.0f + __expf(-gate[0]));
        gv = gg; gc = 1.0f - gg;
    }
    // C/D layout: VGPR j -> M = j + 8*g, N = mrow
    #pragma unroll
    for (int r = 0; r < 4; ++r) {
        #pragma unroll
        for (int c = 0; c < 2; ++c) {
            int col = wave * 32 + c * 16 + mrow;
            #pragma unroll
            for (int j = 0; j < 8; ++j) {
                int rr = row0 + r * 16 + j + 8 * g;
                if (rr < N) {
                    float y = acc[r][c][j];
                    if (bias) y += bias[col];
                    size_t o = (size_t)rr * 256 + col;
                    if (resid) y = gv * y + gc * resid[o];
                    Y[o] = y;
                }
            }
        }
    }
}

// ------------- per-(node,head) dot with half an attention row -------------
__global__ void rowdot8(const float* __restrict__ F, const float* __restrict__ attn,
                        int attnOff, float* __restrict__ out, int n)
{
    int idx = blockIdx.x * blockDim.x + threadIdx.x;
    if (idx >= n * KHEADS) return;
    int node = idx >> 3, k = idx & 7;
    const float* fr = F + (size_t)node * KD + k * DHEAD;
    const float* av = attn + k * 64 + attnOff;
    float s = 0.0f;
    #pragma unroll
    for (int d = 0; d < DHEAD; ++d) s += fr[d] * av[d];
    out[idx] = s;
}

__global__ void fillf(float* __restrict__ p, float v, size_t n)
{
    size_t i = (size_t)blockIdx.x * blockDim.x + threadIdx.x;
    if (i < n) p[i] = v;
}

// -------- edge pass 1: per-dst segment max of leaky_relu(el+er) --------
__global__ void edge_max(const int* __restrict__ src, const int* __restrict__ dst,
                         const float* __restrict__ el, const float* __restrict__ er,
                         float* __restrict__ m)
{
    int idx = blockIdx.x * blockDim.x + threadIdx.x;
    if (idx >= NEDGE * KHEADS) return;
    int e = idx >> 3, k = idx & 7;
    float sc = lrelu(el[src[e] * KHEADS + k] + er[dst[e] * KHEADS + k]);
    int* addr = (int*)(m + (size_t)dst[e] * KHEADS + k);
    int cur = *addr;
    while (__int_as_float(cur) < sc) {
        int prev = atomicCAS(addr, cur, __float_as_int(sc));
        if (prev == cur) break;
        cur = prev;
    }
}

// -------- edge pass 2: ex = exp(score - m[dst]); s[dst] += ex --------
__global__ void edge_exp(const int* __restrict__ src, const int* __restrict__ dst,
                         const float* __restrict__ el, const float* __restrict__ er,
                         const float* __restrict__ m, float* __restrict__ s,
                         float* __restrict__ ex)
{
    int idx = blockIdx.x * blockDim.x + threadIdx.x;
    if (idx >= NEDGE * KHEADS) return;
    int e = idx >> 3, k = idx & 7;
    int dn = dst[e];
    float sc = lrelu(el[src[e] * KHEADS + k] + er[dn * KHEADS + k]);
    float v = __expf(sc - m[(size_t)dn * KHEADS + k]);
    ex[idx] = v;
    atomicAdd(&s[(size_t)dn * KHEADS + k], v);
}

// -------- edge pass 3: ft[dst] += fs[src] * (ex / s[dst]) --------
__global__ __launch_bounds__(256) void edge_agg(
    const int* __restrict__ src, const int* __restrict__ dst,
    const float* __restrict__ ex, const float* __restrict__ s,
    const float* __restrict__ fs, float* __restrict__ ft)
{
    int t = threadIdx.x;       // 0..255 => head k = t>>5, dim d = t&31
    int k = t >> 5;
    for (int e = blockIdx.x; e < NEDGE; e += gridDim.x) {
        int sn = src[e], dn = dst[e];
        float a = ex[(size_t)e * KHEADS + k] / s[(size_t)dn * KHEADS + k];
        float v = fs[(size_t)sn * KD + t] * a;
        atomicAdd(&ft[(size_t)dn * KD + t], v);
    }
}

// -------- macro semantic-attention fusion for papers (+gated residual) ----
// outp currently holds res_paper; overwritten with fused result.
__global__ void paper_fuse(const float* __restrict__ nodep, const float* __restrict__ rfw,
                           const float* __restrict__ rfc, const float* __restrict__ attn,
                           const float* __restrict__ gate, float* __restrict__ outp)
{
    int idx = blockIdx.x * blockDim.x + threadIdx.x;
    if (idx >= N_PAPER * KHEADS) return;
    int n = idx >> 3, k = idx & 7;
    const float* np = nodep + (size_t)n * KD + k * DHEAD;
    const float* rw = rfw   + (size_t)n * KD + k * DHEAD;
    const float* rc = rfc   + (size_t)n * KD + k * DHEAD;
    const float* aL = attn + k * 64;
    const float* aR = aL + DHEAD;
    float sw = 0.0f, sc = 0.0f;
    #pragma unroll
    for (int d = 0; d < DHEAD; ++d) {
        float nv = aL[d] * np[d];
        sw += nv + aR[d] * rw[d];
        sc += nv + aR[d] * rc[d];
    }
    sw = lrelu(sw); sc = lrelu(sc);
    float mx = fmaxf(sw, sc);
    float e1 = __expf(sw - mx), e2 = __expf(sc - mx);
    float inv = 1.0f / (e1 + e2);
    e1 *= inv; e2 *= inv;
    float gp = 1.0f / (1.0f + __expf(-gate[0]));
    float* op = outp + (size_t)n * KD + k * DHEAD;
    #pragma unroll
    for (int d = 0; d < DHEAD; ++d) {
        float o = e1 * rw[d] + e2 * rc[d];
        op[d] = gp * o + (1.0f - gp) * op[d];
    }
}

// =====================================================================
extern "C" void kernel_launch(void* const* d_in, const int* in_sizes, int n_in,
                              void* d_out, int out_size, void* d_ws, size_t ws_size,
                              hipStream_t stream)
{
    const float* fa    = (const float*)d_in[0];
    const float* fp    = (const float*)d_in[1];
    const float* Wma   = (const float*)d_in[2];
    const float* Wmp   = (const float*)d_in[3];
    const float* attA  = (const float*)d_in[4];
    const float* attP  = (const float*)d_in[5];
    const float* Wnp   = (const float*)d_in[7];   // W_macro_node_author (d_in[6]) unused
    const float* Wrw   = (const float*)d_in[8];
    const float* Wrwb  = (const float*)d_in[9];
    const float* Wrc   = (const float*)d_in[10];
    const float* attM  = (const float*)d_in[11];
    const float* Wresa = (const float*)d_in[12];
    const float* bresa = (const float*)d_in[13];
    const float* Wresp = (const float*)d_in[14];
    const float* bresp = (const float*)d_in[15];
    const float* gwa   = (const float*)d_in[16];
    const float* gwp   = (const float*)d_in[17];
    const int*   wsrc  = (const int*)d_in[18];
    const int*   wdstp = (const int*)d_in[19];
    const int*   wbsrc = (const int*)d_in[20];
    const int*   wbdst = (const int*)d_in[21];
    const int*   csrc  = (const int*)d_in[22];
    const int*   cdst  = (const int*)d_in[23];

    float* out_a = (float*)d_out;                       // 50000 x 256
    float* out_p = out_a + (size_t)N_AUTHOR * KD;       // 50000 x 256

    const size_t NF = (size_t)N_AUTHOR * KD;            // 12.8M floats
    float* w = (float*)d_ws;
    float* A_micro = w; w += NF;
    float* P_micro = w; w += NF;
    float* node_p  = w; w += NF;
    float* rf_w    = w; w += NF;
    float* rf_c    = w; w += NF;
    float* ft_w    = w; w += NF;    // papers  (writes)
    float* ft_wb   = w; w += NF;    // authors (written_by)
    float* ft_c    = w; w += NF;    // papers  (cites)
    float* elA_a   = w; w += (size_t)N_AUTHOR * KHEADS;
    float* erP_a   = w; w += (size_t)N_PAPER  * KHEADS;
    float* elP_p   = w; w += (size_t)N_PAPER  * KHEADS;
    float* erA_p   = w; w += (size_t)N_AUTHOR * KHEADS;
    float* erP_p   = w; w += (size_t)N_PAPER  * KHEADS;
    float* m_w     = w; w += (size_t)N_PAPER  * KHEADS; // segment max / sum
    float* m_wb    = w; w += (size_t)N_AUTHOR * KHEADS;
    float* m_c     = w; w += (size_t)N_PAPER  * KHEADS;
    float* s_w     = w; w += (size_t)N_PAPER  * KHEADS;
    float* s_wb    = w; w += (size_t)N_AUTHOR * KHEADS;
    float* s_c     = w; w += (size_t)N_PAPER  * KHEADS;
    float* ex_w    = w; w += (size_t)NEDGE * KHEADS;
    float* ex_wb   = w; w += (size_t)NEDGE * KHEADS;
    float* ex_c    = w; w += (size_t)NEDGE * KHEADS;

    const dim3 blk(256);
    const dim3 gemmGrid((N_AUTHOR + 63) / 64);
    const dim3 nkGrid((N_AUTHOR * KHEADS + 255) / 256);
    const dim3 ekGrid((NEDGE * KHEADS + 255) / 256);
    const dim3 aggGrid(65536);

    // ---- projections (bf16 WMMA GEMMs) ----
    gemm_n256<<<gemmGrid, blk, 0, stream>>>(fa, Wma,  nullptr, nullptr, nullptr, A_micro, N_AUTHOR, 0);
    gemm_n256<<<gemmGrid, blk, 0, stream>>>(fp, Wmp,  nullptr, nullptr, nullptr, P_micro, N_PAPER, 0);
    gemm_n256<<<gemmGrid, blk, 0, stream>>>(fa, Wresa, bresa,  nullptr, nullptr, out_a,   N_AUTHOR, 0);
    gemm_n256<<<gemmGrid, blk, 0, stream>>>(fp, Wresp, bresp,  nullptr, nullptr, out_p,   N_PAPER, 0);
    gemm_n256<<<gemmGrid, blk, 0, stream>>>(fp, Wnp,  nullptr, nullptr, nullptr, node_p,  N_PAPER, 0);

    // ---- per-head attention logits (el/er), keyed by source node type ----
    rowdot8<<<nkGrid, blk, 0, stream>>>(A_micro, attA, 0,     elA_a, N_AUTHOR);
    rowdot8<<<nkGrid, blk, 0, stream>>>(P_micro, attA, DHEAD, erP_a, N_PAPER);
    rowdot8<<<nkGrid, blk, 0, stream>>>(P_micro, attP, 0,     elP_p, N_PAPER);
    rowdot8<<<nkGrid, blk, 0, stream>>>(A_micro, attP, DHEAD, erA_p, N_AUTHOR);
    rowdot8<<<nkGrid, blk, 0, stream>>>(P_micro, attP, DHEAD, erP_p, N_PAPER);

    // ---- init accumulators (m/s/ft regions are contiguous in ws) ----
    {
        size_t nm = (size_t)(N_PAPER + N_AUTHOR + N_PAPER) * KHEADS;
        fillf<<<dim3((nm + 255) / 256), blk, 0, stream>>>(m_w, -1e30f, nm);
        fillf<<<dim3((nm + 255) / 256), blk, 0, stream>>>(s_w, 0.0f, nm);
        size_t nft = 3 * NF;
        fillf<<<dim3((unsigned)((nft + 255) / 256)), blk, 0, stream>>>(ft_w, 0.0f, nft);
    }

    // ---- relation: author -writes-> paper (attn_micro_author) ----
    edge_max<<<ekGrid, blk, 0, stream>>>(wsrc, wdstp, elA_a, erP_a, m_w);
    edge_exp<<<ekGrid, blk, 0, stream>>>(wsrc, wdstp, elA_a, erP_a, m_w, s_w, ex_w);
    edge_agg<<<aggGrid, blk, 0, stream>>>(wsrc, wdstp, ex_w, s_w, A_micro, ft_w);

    // ---- relation: paper -written_by-> author (attn_micro_paper) ----
    edge_max<<<ekGrid, blk, 0, stream>>>(wbsrc, wbdst, elP_p, erA_p, m_wb);
    edge_exp<<<ekGrid, blk, 0, stream>>>(wbsrc, wbdst, elP_p, erA_p, m_wb, s_wb, ex_wb);
    edge_agg<<<aggGrid, blk, 0, stream>>>(wbsrc, wbdst, ex_wb, s_wb, P_micro, ft_wb);

    // ---- relation: paper -cites-> paper (attn_micro_paper) ----
    edge_max<<<ekGrid, blk, 0, stream>>>(csrc, cdst, elP_p, erP_p, m_c);
    edge_exp<<<ekGrid, blk, 0, stream>>>(csrc, cdst, elP_p, erP_p, m_c, s_c, ex_c);
    edge_agg<<<aggGrid, blk, 0, stream>>>(csrc, cdst, ex_c, s_c, P_micro, ft_c);

    // ---- macro relation GEMMs over relu(ft) ----
    // author: single relation -> gated blend with residual, in-place on out_a
    gemm_n256<<<gemmGrid, blk, 0, stream>>>(ft_wb, Wrwb, nullptr, gwa, out_a, out_a, N_AUTHOR, 1);
    gemm_n256<<<gemmGrid, blk, 0, stream>>>(ft_w,  Wrw,  nullptr, nullptr, nullptr, rf_w, N_PAPER, 1);
    gemm_n256<<<gemmGrid, blk, 0, stream>>>(ft_c,  Wrc,  nullptr, nullptr, nullptr, rf_c, N_PAPER, 1);

    // ---- paper: semantic attention over {writes, cites} + gated residual ----
    paper_fuse<<<nkGrid, blk, 0, stream>>>(node_p, rf_w, rf_c, attM, gwp, out_p);

    (void)in_sizes; (void)n_in; (void)out_size; (void)ws_size;
}
